// GMN_Encoder_58798102282930
// MI455X (gfx1250) — compile-verified, matrix operand order checked by reading, not run on hardware
//
#include <hip/hip_runtime.h>
#include <hip/hip_bf16.h>
#include <math.h>

#define DD 128
#define NN 512
#define BB 512
#define ROWS 64   // rows per GEMM block

typedef __attribute__((ext_vector_type(16))) _Float16 v16h;
typedef __attribute__((ext_vector_type(8)))  float    v8f;

struct __align__(8)  h4 { _Float16 v[4]; };
struct __align__(16) h8 { _Float16 v[8]; };
union V16 { v16h v; h8 h[2]; };

__device__ __forceinline__ float lrelu(float x) {
  return x > 0.0f ? x : 0.01f * x;
}

// ---------------------------------------------------------------------------
// Kernel 0: tiny weight prep. Writes weights TRANSPOSED ([col][k]) so the
// GEMM kernel's B-fragment LDS reads are contiguous 16B loads.
//   WgT[j,i] = (Wk @ Wa1)[i,j]   (Wa1 = W_att rows [0,128))
//   WvT[j,i] = Wv[i,j]
// ---------------------------------------------------------------------------
__global__ __launch_bounds__(256) void prep_weights(
    const float* __restrict__ Wk, const float* __restrict__ Wv,
    const float* __restrict__ Watt,
    _Float16* __restrict__ WgT, _Float16* __restrict__ WvT) {
  int idx = blockIdx.x * 256 + threadIdx.x;
  if (idx < DD * DD) {
    int i = idx >> 7, j = idx & 127;
    float s = 0.0f;
    #pragma unroll 4
    for (int k = 0; k < DD; ++k) s += Wk[i * DD + k] * Watt[k * DD + j];
    WgT[j * DD + i] = (_Float16)s;
    WvT[j * DD + i] = (_Float16)Wv[i * DD + j];
  }
}

// ---------------------------------------------------------------------------
// Kernel 1: Y[r,:] = (X[r,:] * mask[r]) @ W   (f32 in, f16 out), WMMA f16.
// Block: 256 thr = 8 waves; block owns 64 rows; wave w owns cols [16w,16w+16)
// for all 4 row-tiles (16 x v_wmma_f32_16x16x32_f16 per wave).
// Outputs staged through LDS (reusing sA) for coalesced b128 global stores.
// ---------------------------------------------------------------------------
__global__ __launch_bounds__(256) void gemm_masked_f16(
    const float* __restrict__ X, const int* __restrict__ mask,
    const _Float16* __restrict__ WT16, _Float16* __restrict__ Y) {
  __shared__ _Float16 sW[DD][DD + 8];    // [col][k], transposed weight
  __shared__ _Float16 sA[ROWS][DD + 8];  // [row][k]; reused as output tile
  const int t = threadIdx.x;
  const int row0 = blockIdx.x * ROWS;

  // Stage transposed weight: 16B copies.
  for (int i = t * 8; i < DD * DD; i += 2048)
    *(h8*)&sW[i >> 7][i & 127] = *(const h8*)&WT16[i];
  // Stage masked A tile: float4 global loads -> h4 LDS stores.
  for (int i = t * 4; i < ROWS * DD; i += 1024) {
    const int r = i >> 7, c = i & 127;
    const float mval = (float)mask[row0 + r];
    const float4 x = *(const float4*)&X[(size_t)(row0 + r) * DD + c];
    h4 hh;
    hh.v[0] = (_Float16)(x.x * mval);
    hh.v[1] = (_Float16)(x.y * mval);
    hh.v[2] = (_Float16)(x.z * mval);
    hh.v[3] = (_Float16)(x.w * mval);
    *(h4*)&sA[r][c] = hh;
  }
  __syncthreads();

  const int wave = t >> 5, lane = t & 31;
  const int lo = lane & 15;
  const bool hi = lane >= 16;
  const int hioff = hi ? 8 : 0;
  const int ncol = wave * 16 + lo;

  // ISA 7.12.2 16-bit fragment layouts (wave32): halves 0-7 and 8-15 are two
  // contiguous K-octets at kb+hioff and kb+hioff+16 -> two b128 LDS loads.
  V16 bfr[4];
  #pragma unroll
  for (int ks = 0; ks < 4; ++ks) {
    const int kb = ks * 32;
    bfr[ks].h[0] = *(const h8*)&sW[ncol][kb + hioff];
    bfr[ks].h[1] = *(const h8*)&sW[ncol][kb + hioff + 16];
  }

  v8f accs[4];
  #pragma unroll
  for (int rt = 0; rt < 4; ++rt) {
    const int r0 = rt * 16;
    v8f acc = {};
    #pragma unroll
    for (int ks = 0; ks < 4; ++ks) {
      const int kb = ks * 32;
      V16 afr;
      afr.h[0] = *(const h8*)&sA[r0 + lo][kb + hioff];
      afr.h[1] = *(const h8*)&sA[r0 + lo][kb + hioff + 16];
      acc = __builtin_amdgcn_wmma_f32_16x16x32_f16(false, afr.v, false, bfr[ks].v,
                                                   (short)0, acc, false, false);
    }
    accs[rt] = acc;
  }

  // All WMMA reads of sA are done; reuse it as the f16 output tile.
  __syncthreads();
  #pragma unroll
  for (int rt = 0; rt < 4; ++rt) {
    #pragma unroll
    for (int r8 = 0; r8 < 8; ++r8) {
      // C/D layout: VGPR r8 -> M = r8 (+8 for lanes 16-31), N = lane&15.
      const int r = rt * 16 + (hi ? (r8 + 8) : r8);
      sA[r][ncol] = (_Float16)accs[rt][r8];
    }
  }
  __syncthreads();

  // Coalesced output: 16B chunks, 16 threads per 256B row.
  for (int i = t * 8; i < ROWS * DD; i += 2048) {
    const int r = i >> 7, c = i & 127;
    *(h8*)&Y[(size_t)(row0 + r) * DD + c] = *(const h8*)&sA[r][c];
  }
}

// ---------------------------------------------------------------------------
// Kernel 2: per-b 3-hop loop (rows are independent across hops).
//   C[d]   = u@Wa2 + rel@Wa3 + b_att
//   att[n] = sum_d lrelu(G16[b,n,d] + C[d]);  p = masked softmax
//   o[d]   = sum_n p[n] * fv16[b,n,d]
//   u      = renorm_1e-12(lrelu(u@Wlin+blin) + o)
// G/fv streamed with 16B loads: wave covers 2 rows x 128 d per iteration.
// ---------------------------------------------------------------------------
__global__ __launch_bounds__(256) void hops_kernel(
    const float* __restrict__ e1, const float* __restrict__ rel,
    const int* __restrict__ mask,
    const float* __restrict__ Wlin, const float* __restrict__ blin,
    const float* __restrict__ Watt, const float* __restrict__ batt,
    const _Float16* __restrict__ G16, const _Float16* __restrict__ FV16,
    float* __restrict__ out) {
  __shared__ float su[DD], slin[DD], sC[DD], srb[DD], so[DD];
  __shared__ float satt[NN];
  __shared__ float red[256];
  __shared__ float osum[16 * DD];

  const int t = threadIdx.x;
  const int b = blockIdx.x;
  const int wave = t >> 5, lane = t & 31;
  const int rowoff = lane >> 4;          // 0 or 1: second row of the pair
  const int dbase = (lane & 15) * 8;     // this lane's d-octet
  const _Float16* Gb = G16 + (size_t)b * NN * DD;
  const _Float16* Fb = FV16 + (size_t)b * NN * DD;

  if (t < DD) {
    su[t] = e1[b * DD + t];
    float s = 0.0f;
    for (int k = 0; k < DD; ++k) s += rel[b * DD + k] * Watt[(2 * DD + k) * DD + t];
    srb[t] = s;  // rel @ Wa3, hop-invariant
  }
  __syncthreads();

  for (int hop = 0; hop < 3; ++hop) {
    // u_batch (after lrelu) and C vector from current u.
    if (t < DD) {
      float s1 = blin[t];
      float s2 = batt[t] + srb[t];
      for (int k = 0; k < DD; ++k) {
        const float uk = su[k];
        s1 += uk * Wlin[k * DD + t];
        s2 += uk * Watt[(DD + k) * DD + t];
      }
      slin[t] = lrelu(s1);
      sC[t] = s2;
    }
    __syncthreads();

    // att[n] = sum_d lrelu(G + C): each wave handles 2 rows per iteration.
    for (int n0 = wave * 2; n0 < NN; n0 += 16) {
      const int n = n0 + rowoff;
      h8 g = *(const h8*)(Gb + (size_t)n * DD + dbase);
      __builtin_prefetch(Fb + (size_t)n * DD + dbase, 0, 0);  // warm fv for o-pass
      float s = 0.0f;
      #pragma unroll
      for (int i = 0; i < 8; ++i) s += lrelu((float)g.v[i] + sC[dbase + i]);
      #pragma unroll
      for (int off = 1; off <= 8; off <<= 1) s += __shfl_xor(s, off, 32);
      if ((lane & 15) == 0) satt[n] = s;
    }
    __syncthreads();

    // masked softmax over n (max over all n, mask applied to exp, +1e-5).
    float lmax = -3.4e38f;
    for (int n = t; n < NN; n += 256) lmax = fmaxf(lmax, satt[n]);
    red[t] = lmax;
    __syncthreads();
    for (int s = 128; s > 0; s >>= 1) {
      if (t < s) red[t] = fmaxf(red[t], red[t + s]);
      __syncthreads();
    }
    const float mx = red[0];
    __syncthreads();

    float lsum = 0.0f;
    for (int n = t; n < NN; n += 256) {
      const float e = expf(satt[n] - mx) * (float)mask[b * NN + n];
      satt[n] = e;
      lsum += e;
    }
    red[t] = lsum;
    __syncthreads();
    for (int s = 128; s > 0; s >>= 1) {
      if (t < s) red[t] += red[t + s];
      __syncthreads();
    }
    const float inv = 1.0f / (red[0] + 1e-5f);
    __syncthreads();
    for (int n = t; n < NN; n += 256) satt[n] *= inv;
    __syncthreads();

    // o[d] = sum_n p[n]*fv[n,d]: same 2-rows-per-wave streaming.
    float a[8];
    #pragma unroll
    for (int i = 0; i < 8; ++i) a[i] = 0.0f;
    for (int n0 = wave * 2; n0 < NN; n0 += 16) {
      const int n = n0 + rowoff;
      const float pn = satt[n];
      h8 g = *(const h8*)(Fb + (size_t)n * DD + dbase);
      #pragma unroll
      for (int i = 0; i < 8; ++i) a[i] += pn * (float)g.v[i];
    }
    {
      const int slot = wave * 2 + rowoff;  // 0..15
      #pragma unroll
      for (int i = 0; i < 8; ++i) osum[slot * DD + dbase + i] = a[i];
    }
    __syncthreads();

    if (t < DD) {
      float s = 0.0f;
      #pragma unroll
      for (int w = 0; w < 16; ++w) s += osum[w * DD + t];
      so[t] = s;
      const float v = slin[t] + s;
      red[t] = v * v;
    } else {
      red[t] = 0.0f;
    }
    __syncthreads();
    for (int s = 128; s > 0; s >>= 1) {
      if (t < s) red[t] += red[t + s];
      __syncthreads();
    }
    const float nrm = sqrtf(red[0]);
    __syncthreads();
    if (t < DD) {
      const float v = slin[t] + so[t];
      su[t] = (nrm > 1e-12f) ? (v / nrm) : (v * 1e12f);
    }
    __syncthreads();
  }

  if (t < DD) out[b * DD + t] = su[t];
}

// ---------------------------------------------------------------------------
extern "C" void kernel_launch(void* const* d_in, const int* in_sizes, int n_in,
                              void* d_out, int out_size, void* d_ws, size_t ws_size,
                              hipStream_t stream) {
  (void)in_sizes; (void)n_in; (void)out_size; (void)ws_size;
  const float* e1   = (const float*)d_in[0];
  const float* rel  = (const float*)d_in[1];
  const float* key  = (const float*)d_in[2];
  const float* val  = (const float*)d_in[3];
  const int*   mask = (const int*)d_in[4];
  const float* Wk   = (const float*)d_in[5];
  const float* Wv   = (const float*)d_in[6];
  const float* Wlin = (const float*)d_in[7];
  const float* blin = (const float*)d_in[8];
  const float* Watt = (const float*)d_in[9];
  const float* batt = (const float*)d_in[10];
  float* out = (float*)d_out;

  // Workspace: G16 (64MB) | FV16 (64MB) | WgT (32KB) | WvT (32KB)
  _Float16* G16 = (_Float16*)d_ws;
  _Float16* FV16 = G16 + (size_t)BB * NN * DD;
  _Float16* WgT = FV16 + (size_t)BB * NN * DD;
  _Float16* WvT = WgT + DD * DD;

  prep_weights<<<64, 256, 0, stream>>>(Wk, Wv, Watt, WgT, WvT);
  const int R = BB * NN;
  gemm_masked_f16<<<R / ROWS, 256, 0, stream>>>(key, mask, WgT, G16);
  gemm_masked_f16<<<R / ROWS, 256, 0, stream>>>(val, mask, WvT, FV16);
  hops_kernel<<<BB, 256, 0, stream>>>(e1, rel, mask, Wlin, blin, Watt, batt,
                                      G16, FV16, out);
}
